// DistanceAwareMultiheadAttention_34170759807100
// MI455X (gfx1250) — matrix operand order repaired and checked by reading, not compiled, not run on hardware
//
#include <hip/hip_runtime.h>
#include <hip/hip_bf16.h>
#include <math.h>

// ---------------------------------------------------------------------------
// DistanceAwareMultiheadAttention forward for MI455X (gfx1250, wave32, WMMA).
// N=1536 nodes, D=512, H=8 heads, DH=64.
// ---------------------------------------------------------------------------

#define NN 1536
#define DD 512
#define HH 8
#define DHD 64
#define SCALE_REL 362.03867196751236f   // 256*sqrt(2)
#define INV_SQRT_DH 0.125f
#define DIVMUL 0.74989420613f           // exp(-2*ln(10000)/64)

typedef __attribute__((ext_vector_type(16))) _Float16 v16h;
typedef __attribute__((ext_vector_type(8)))  float    v8f;

union U16h { unsigned u[8]; v16h v; };

// LDS byte offset of a shared-memory object (generic LDS addr: low 32 bits).
__device__ __forceinline__ unsigned lds_off(const void* p) {
    return (unsigned)(uintptr_t)p;
}

// Async 16B global -> LDS copy (ASYNCcnt-tracked, gfx1250).
__device__ __forceinline__ void async_g2l_b128(unsigned lds, const void* g) {
    asm volatile("global_load_async_to_lds_b128 %0, %1, off"
                 :: "v"(lds), "v"((unsigned long long)(uintptr_t)g)
                 : "memory");
}

// A-operand (16x32 f16) loader per ISA layout:
//   lane m = lane&15 (both halves), VGPR j holds K pair at
//   (j>>2)*16 + half*8 + (j&3)*2
__device__ __forceinline__ v16h load_A(const _Float16* base, int stride,
                                       int m, int k0, int half) {
    U16h r;
    const _Float16* p = base + m * stride + k0;
#pragma unroll
    for (int j = 0; j < 8; ++j) {
        int kp = ((j >> 2) << 4) + (half << 3) + ((j & 3) << 1);
        r.u[j] = *(const unsigned*)(p + kp);
    }
    return r.v;
}

// B-operand (32x16 f16) loader per ISA layout:
//   lane n = lane&15, VGPR j holds K pair at (j>>2)*16 + half*8 + (j&3)*2
__device__ __forceinline__ v16h load_B(const _Float16* base, int stride,
                                       int coln, int k0, int half) {
    U16h r;
    const _Float16* p = base + coln * stride + k0;
#pragma unroll
    for (int j = 0; j < 8; ++j) {
        int kp = ((j >> 2) << 4) + (half << 3) + ((j & 3) << 1);
        r.u[j] = *(const unsigned*)(p + kp);
    }
    return r.v;
}

// ---------------------------------------------------------------------------
// Kernel 1: QKV projection.  out = features @ W^T + b, f16 out in [H][N][64].
// block = 128 threads (4 waves); block tile = 64 rows x 64 cols; z selects Q/K/V.
// ---------------------------------------------------------------------------
__global__ __launch_bounds__(128) void qkv_proj_kernel(
    const float* __restrict__ feat,
    const float* __restrict__ Wq, const float* __restrict__ bq,
    const float* __restrict__ Wk, const float* __restrict__ bk,
    const float* __restrict__ Wv, const float* __restrict__ bv,
    _Float16* __restrict__ qf, _Float16* __restrict__ kf,
    _Float16* __restrict__ vf) {
    const float* W;
    const float* bias;
    _Float16* out;
    if (blockIdx.z == 0)      { W = Wq; bias = bq; out = qf; }
    else if (blockIdx.z == 1) { W = Wk; bias = bk; out = kf; }
    else                      { W = Wv; bias = bv; out = vf; }

    __shared__ __align__(16) _Float16 As[64][72];
    __shared__ __align__(16) _Float16 Bs[64][72];

    const int tid = threadIdx.x;
    const int w = tid >> 5, lane = tid & 31;
    const int half = lane >> 4, col = lane & 15;
    const int n0 = blockIdx.x * 64, j0 = blockIdx.y * 64;

    v8f acc[4] = {};

    for (int kc = 0; kc < 8; ++kc) {
        const int k0 = kc * 64;
        // cooperative f32 -> f16 staged loads (coalesced in c)
        for (int t = 0; t < 32; ++t) {
            int e = t * 128 + tid;
            int r = e >> 6, c = e & 63;
            As[r][c] = (_Float16)feat[(size_t)(n0 + r) * DD + k0 + c];
            Bs[r][c] = (_Float16)W[(size_t)(j0 + r) * DD + k0 + c];
        }
        __syncthreads();
#pragma unroll
        for (int ks = 0; ks < 2; ++ks) {
            v16h a = load_A(&As[0][0], 72, w * 16 + col, ks * 32, half);
#pragma unroll
            for (int nb = 0; nb < 4; ++nb) {
                v16h b = load_B(&Bs[0][0], 72, nb * 16 + col, ks * 32, half);
                acc[nb] = __builtin_amdgcn_wmma_f32_16x16x32_f16(
                    false, a, false, b, (short)0, acc[nb], false, false);
            }
        }
        __syncthreads();
    }

    // epilogue: bias + convert + store to [h][n][dh] (head-split of column)
#pragma unroll
    for (int nb = 0; nb < 4; ++nb) {
        int jcol = j0 + nb * 16 + col;
        float bb = bias[jcol];
        int hh = jcol >> 6, dh = jcol & 63;
#pragma unroll
        for (int j = 0; j < 8; ++j) {
            int n = n0 + w * 16 + j + 8 * half;  // C-layout row mapping
            out[((size_t)hh * NN + n) * DHD + dh] = (_Float16)(acc[nb][j] + bb);
        }
    }
}

// ---------------------------------------------------------------------------
// Kernel 2: per-row sums of odd elements of q and k (rk base, rel==0 path).
// ---------------------------------------------------------------------------
__global__ void oddsum_kernel(const _Float16* __restrict__ qf,
                              const _Float16* __restrict__ kf,
                              float* __restrict__ qo, float* __restrict__ ko) {
    int i = blockIdx.x * blockDim.x + threadIdx.x;
    if (i < HH * NN) {
        const _Float16* qr = qf + (size_t)i * DHD;
        const _Float16* kr = kf + (size_t)i * DHD;
        float sq = 0.f, sk = 0.f;
#pragma unroll 8
        for (int t = 1; t < DHD; t += 2) {
            sq += (float)qr[t];
            sk += (float)kr[t];
        }
        qo[i] = sq;
        ko[i] = sk;
    }
}

// ---------------------------------------------------------------------------
// Kernel 3a/3b: build dense rel = scatter_add(edge_attr) * SCALE
// ---------------------------------------------------------------------------
__global__ void zero_f32_kernel(float* __restrict__ p, int n) {
    int i = blockIdx.x * blockDim.x + threadIdx.x;
    int stride = gridDim.x * blockDim.x;
    for (; i < n; i += stride) p[i] = 0.f;
}

__global__ void scatter_rel_kernel(const int* __restrict__ ei,
                                   const float* __restrict__ ea,
                                   float* __restrict__ rel, int E) {
    int e = blockIdx.x * blockDim.x + threadIdx.x;
    if (e < E) {
        int s = ei[e];
        int d = ei[E + e];
        atomicAdd(&rel[(size_t)s * NN + d], ea[e] * SCALE_REL);
    }
}

// ---------------------------------------------------------------------------
// Kernel 4: fused flash attention with sparse rel-PE correction.
// grid = (N/64, H); block = 128 threads = 4 waves; wave owns 16 query rows.
// K tile / rel tile / Q tile arrive via async global->LDS (ASYNCcnt path);
// V tile is transposed in flight (B-operand layout), so it stays manual.
// ---------------------------------------------------------------------------
__global__ __launch_bounds__(128) void attn_kernel(
    const _Float16* __restrict__ qf, const _Float16* __restrict__ kf,
    const _Float16* __restrict__ vf, const float* __restrict__ rel,
    const float* __restrict__ qo, const float* __restrict__ ko,
    float* __restrict__ attn_out) {
    // 16B-aligned LDS rows (async B128 destination requirement)
    __shared__ __align__(16) _Float16 Qs[4][16][80];   // per-wave Q tiles
    __shared__ __align__(16) _Float16 Ks[32][80];      // shared K tile (32 keys)
    __shared__ __align__(16) _Float16 Vt[64][40];      // V transposed [d][key]
    __shared__ __align__(16) _Float16 Ps[4][16][40];   // per-wave P scratch
    __shared__ __align__(16) float Rels[64][40];       // rel tile [qlocal][key]

    const int tid = threadIdx.x;
    const int w = tid >> 5, lane = tid & 31;
    const int half = lane >> 4, col = lane & 15;
    const int h = blockIdx.y;
    const int qg0 = blockIdx.x * 64;   // block's first query row
    const int q0w = qg0 + w * 16;      // wave's first query row

    // async-load this wave's 16x64 Q tile into LDS (16 rows x 128B)
    const _Float16* qbase = qf + ((size_t)h * NN + q0w) * DHD;
#pragma unroll
    for (int t = 0; t < 4; ++t) {
        int c = t * 32 + lane;            // 16B chunk id; 8 chunks per row
        int r = c >> 3, dc = (c & 7) * 8;
        async_g2l_b128(lds_off(&Qs[w][r][dc]), qbase + r * DHD + dc);
    }

    float m_i[8], l_i[8], qov[8];
    v8f O[4] = {};
#pragma unroll
    for (int j = 0; j < 8; ++j) {
        m_i[j] = -3.0e38f;
        l_i[j] = 0.f;
        qov[j] = qo[h * NN + q0w + j + 8 * half];
    }

    for (int rt = 0; rt < NN / 32; ++rt) {
        const int r0 = rt * 32;
        const _Float16* kb = kf + ((size_t)h * NN + r0) * DHD;
        const _Float16* vb = vf + ((size_t)h * NN + r0) * DHD;
        const float* relb = rel + (size_t)qg0 * NN + r0;
        // prefetch next tiles into L2 while we fill LDS
        __builtin_prefetch(kb + 32 * DHD, 0, 0);
        __builtin_prefetch(vb + 32 * DHD, 0, 0);

        // async K tile: 32 rows x 128B = 256 chunks / 128 threads
#pragma unroll
        for (int t = 0; t < 2; ++t) {
            int c = t * 128 + tid;
            int kk = c >> 3, dc = (c & 7) * 8;
            async_g2l_b128(lds_off(&Ks[kk][dc]), kb + kk * DHD + dc);
        }
        // async rel tile: 64 rows x 128B = 512 chunks / 128 threads
#pragma unroll
        for (int t = 0; t < 4; ++t) {
            int c = t * 128 + tid;
            int ql = c >> 3, fc = (c & 7) * 4;
            async_g2l_b128(lds_off(&Rels[ql][fc]),
                           relb + (size_t)ql * NN + fc);
        }
        // V tile transposed into LDS (manual: layout change)
        for (int t = 0; t < 16; ++t) {
            int e = t * 128 + tid;
            int kk = e >> 6, d = e & 63;
            Vt[d][kk] = vb[kk * DHD + d];
        }
        asm volatile("s_wait_asynccnt 0x0" ::: "memory");
        __syncthreads();

        float Lg[2][8];
#pragma unroll
        for (int st = 0; st < 2; ++st) {
            const int kb16 = st * 16;
            v8f S = {};
#pragma unroll
            for (int ks = 0; ks < 2; ++ks) {
                v16h a = load_A(&Qs[w][0][0], 80, col, ks * 32, half);
                v16h b = load_B(&Ks[0][0], 80, kb16 + col, ks * 32, half);
                S = __builtin_amdgcn_wmma_f32_16x16x32_f16(
                    false, a, false, b, (short)0, S, false, false);
            }
            const int key_l = kb16 + col;       // key within 32-key tile
            const float kov = ko[h * NN + r0 + key_l];
#pragma unroll
            for (int j = 0; j < 8; ++j) {
                const int qlr = w * 16 + j + 8 * half;  // block-local query row
                float rv = Rels[qlr][key_l];
                float T;
                if (rv != 0.f) {
                    // rare path (~2%): full 32-term sinusoidal dot
                    const _Float16* qrow = &Qs[w][j + 8 * half][0];
                    const _Float16* krow = &Ks[key_l][0];
                    float div = 1.f;
                    T = 0.f;
#pragma unroll 4
                    for (int i = 0; i < 32; ++i) {
                        float ang = rv * div;
                        float sv = __sinf(ang), cv = __cosf(ang);
                        T += ((float)qrow[2 * i] + (float)krow[2 * i]) * sv +
                             ((float)qrow[2 * i + 1] + (float)krow[2 * i + 1]) * cv;
                        div *= DIVMUL;
                    }
                } else {
                    T = qov[j] + kov;  // rel==0: sin=0, cos=1 -> odd sums
                }
                Lg[st][j] = (2.f * S[j] + T) * INV_SQRT_DH;
            }
        }

        // ---- online softmax (row = C-layout: half-wave, width-16 shuffles) --
#pragma unroll
        for (int j = 0; j < 8; ++j) {
            float lm = fmaxf(Lg[0][j], Lg[1][j]);
#pragma unroll
            for (int msk = 8; msk > 0; msk >>= 1)
                lm = fmaxf(lm, __shfl_xor(lm, msk, 16));
            float nm = fmaxf(m_i[j], lm);
            float fs = __expf(m_i[j] - nm);
            m_i[j] = nm;
            float p0 = __expf(Lg[0][j] - nm);
            float p1 = __expf(Lg[1][j] - nm);
            float rs = p0 + p1;
#pragma unroll
            for (int msk = 8; msk > 0; msk >>= 1)
                rs += __shfl_xor(rs, msk, 16);
            l_i[j] = l_i[j] * fs + rs;
            O[0][j] *= fs;
            O[1][j] *= fs;
            O[2][j] *= fs;
            O[3][j] *= fs;
            // C-layout -> LDS (for A-layout reload)
            Ps[w][j + 8 * half][col] = (_Float16)p0;
            Ps[w][j + 8 * half][16 + col] = (_Float16)p1;
        }
        // same-wave LDS RAW: force store->load ordering explicitly
        asm volatile("s_wait_dscnt 0" ::: "memory");

        // ---- O += P x V ----------------------------------------------------
        v16h aP = load_A(&Ps[w][0][0], 40, col, 0, half);
#pragma unroll
        for (int nb = 0; nb < 4; ++nb) {
            v16h bV = load_B(&Vt[0][0], 40, nb * 16 + col, 0, half);
            O[nb] = __builtin_amdgcn_wmma_f32_16x16x32_f16(
                false, aP, false, bV, (short)0, O[nb], false, false);
        }
        __syncthreads();  // protect Ks/Vt/Rels for next iteration
    }

    // epilogue: normalize and store [h][n][dh] f32
    const size_t obase = ((size_t)h * NN + q0w) * DHD;
#pragma unroll
    for (int j = 0; j < 8; ++j) {
        float inv = 1.f / l_i[j];
        size_t rb = obase + (size_t)(j + 8 * half) * DHD;
        attn_out[rb + 0 + col]  = O[0][j] * inv;
        attn_out[rb + 16 + col] = O[1][j] * inv;
        attn_out[rb + 32 + col] = O[2][j] * inv;
        attn_out[rb + 48 + col] = O[3][j] * inv;
    }
}

// ---------------------------------------------------------------------------
// Kernel 5: MIL max-pool over nodes -> out[512] (column = h*64+dh)
// ---------------------------------------------------------------------------
__global__ void maxpool_kernel(const float* __restrict__ attn,
                               float* __restrict__ out) {
    int j = blockIdx.x * blockDim.x + threadIdx.x;
    if (j < DD) {
        int h = j >> 6, dh = j & 63;
        const float* p = attn + (size_t)h * NN * DHD + dh;
        float m = -3.0e38f;
        for (int n = 0; n < NN; ++n) m = fmaxf(m, p[(size_t)n * DHD]);
        out[j] = m;
    }
}

// ---------------------------------------------------------------------------
extern "C" void kernel_launch(void* const* d_in, const int* in_sizes, int n_in,
                              void* d_out, int out_size, void* d_ws,
                              size_t ws_size, hipStream_t stream) {
    const float* features  = (const float*)d_in[0];
    const int*   edge_idx  = (const int*)d_in[1];
    const float* edge_attr = (const float*)d_in[2];
    // d_in[3] = pos (unused by forward)
    const float* Wq = (const float*)d_in[4];
    const float* bq = (const float*)d_in[5];
    const float* Wk = (const float*)d_in[6];
    const float* bk = (const float*)d_in[7];
    const float* Wv = (const float*)d_in[8];
    const float* bv = (const float*)d_in[9];
    float* out = (float*)d_out;
    const int E = in_sizes[2];

    // workspace carving
    _Float16* qf = (_Float16*)d_ws;
    _Float16* kf = qf + (size_t)HH * NN * DHD;
    _Float16* vf = kf + (size_t)HH * NN * DHD;
    float* rel  = (float*)(vf + (size_t)HH * NN * DHD);
    float* qo   = rel + (size_t)NN * NN;
    float* ko   = qo + (size_t)HH * NN;
    float* attn = ko + (size_t)HH * NN;

    // 1) QKV projections (WMMA GEMM), z = {Q,K,V}
    qkv_proj_kernel<<<dim3(NN / 64, DD / 64, 3), 128, 0, stream>>>(
        features, Wq, bq, Wk, bk, Wv, bv, qf, kf, vf);

    // 2) odd-element sums for the rel==0 fast path
    oddsum_kernel<<<(HH * NN + 255) / 256, 256, 0, stream>>>(qf, kf, qo, ko);

    // 3) dense rel = scatter(edge_attr)*SCALE
    zero_f32_kernel<<<1024, 256, 0, stream>>>(rel, NN * NN);
    scatter_rel_kernel<<<(E + 255) / 256, 256, 0, stream>>>(edge_idx, edge_attr,
                                                            rel, E);

    // 4) fused flash attention with sparse PE correction
    attn_kernel<<<dim3(NN / 64, HH), 128, 0, stream>>>(qf, kf, vf, rel, qo, ko,
                                                       attn);

    // 5) max-pool over nodes
    maxpool_kernel<<<(DD + 255) / 256, 256, 0, stream>>>(attn, out);
}
